// Attention_12481174962212
// MI455X (gfx1250) — compile-verified, matrix operand order checked by reading, not run on hardware
//
#include <hip/hip_runtime.h>
#include <hip/hip_bf16.h>

typedef __attribute__((ext_vector_type(16))) _Float16 v16h;
typedef __attribute__((ext_vector_type(8)))  _Float16 h8;
typedef __attribute__((ext_vector_type(2)))  _Float16 h2;
typedef __attribute__((ext_vector_type(8)))  float    v8f;
typedef __attribute__((ext_vector_type(4)))  int      i4;

#define HEAD  64
#define CEMB  768
#define BATCH 16
#define SEQ   2048
#define BT    (BATCH * SEQ)

#if defined(__AMDGCN__) && __has_builtin(__builtin_amdgcn_global_load_async_to_lds_b128)
#define USE_ASYNC_LDS 1
#else
#define USE_ASYNC_LDS 0
#endif

// Pointee types for the async-to-LDS builtin:
//   arg0: i4 addrspace(1)* (global source, non-const)
//   arg1: i4 addrspace(3)* (LDS destination)
typedef __attribute__((address_space(1))) i4 as1_i4;
typedef __attribute__((address_space(3))) i4 as3_i4;

__device__ __forceinline__ void wait_async0() {
#if __has_builtin(__builtin_amdgcn_s_wait_asynccnt)
  __builtin_amdgcn_s_wait_asynccnt(0);
#else
  asm volatile("s_wait_asynccnt 0x0" ::: "memory");
#endif
}
__device__ __forceinline__ void wait_ds0() {
#if __has_builtin(__builtin_amdgcn_s_wait_dscnt)
  __builtin_amdgcn_s_wait_dscnt(0);
#else
  asm volatile("s_wait_dscnt 0x0" ::: "memory");
#endif
}

union UV16 { v16h v; _Float16 h[16]; h2 p[8]; };
union UV8F { v8f v; float f[8]; };

__device__ __forceinline__ v8f wmma16(v16h a, v16h b, v8f c) {
  // D = A(16x32 f16) x B(32x16 f16) + C(16x16 f32)
  return __builtin_amdgcn_wmma_f32_16x16x32_f16(false, a, false, b, (short)0, c,
                                                false, false);
}

// Load an A-style fragment (row-major MxK, 16 rows x 32 K) per ISA 7.12.2:
// lanes 0-15: row=lane, halves 0..7 -> K 0..7, halves 8..15 -> K 16..23
// lanes 16-31: row=lane-16, halves 0..7 -> K 8..15, halves 8..15 -> K 24..31
// Also used for B fragments by passing B^T (N x K row-major).
__device__ __forceinline__ v16h frag_rm(const _Float16* p, int ld, int row, int kb) {
  const int lane = threadIdx.x & 31;
  const int r = lane & 15, hi = lane >> 4;
  const _Float16* rp = p + (size_t)(row + r) * ld + kb;
  const int k0 = hi * 8, k1 = 16 + hi * 8;
  UV16 u;
#pragma unroll
  for (int i = 0; i < 4; ++i) {
    u.p[i]     = *(const h2*)(rp + k0 + 2 * i);
    u.p[4 + i] = *(const h2*)(rp + k1 + 2 * i);
  }
  return u.v;
}

// Same fragment but sourced from fp32 memory (converted on the fly).
__device__ __forceinline__ v16h frag_rm_f32(const float* p, int ld, int row, int kb) {
  const int lane = threadIdx.x & 31;
  const int r = lane & 15, hi = lane >> 4;
  const float* rp = p + (size_t)(row + r) * ld + kb;
  const int k0 = hi * 8, k1 = 16 + hi * 8;
  UV16 u;
#pragma unroll
  for (int i = 0; i < 8; ++i) {
    u.h[i]     = (_Float16)rp[k0 + i];
    u.h[8 + i] = (_Float16)rp[k1 + i];
  }
  return u.v;
}

// Gather a B fragment whose B^T (N x K) is stored K-major in LDS as [K][ld]:
// lane column n = ncol + lane%16, K pattern as above (column gather).
__device__ __forceinline__ v16h frag_colgather(const _Float16* lds, int ld, int ncol) {
  const int lane = threadIdx.x & 31;
  const int n = ncol + (lane & 15);
  const int hi = lane >> 4;
  const int k0 = hi * 8, k1 = 16 + hi * 8;
  UV16 u;
#pragma unroll
  for (int i = 0; i < 8; ++i) {
    u.h[i]     = lds[(k0 + i) * ld + n];
    u.h[8 + i] = lds[(k1 + i) * ld + n];
  }
  return u.v;
}

__device__ __forceinline__ float red_max16(float x) {
#pragma unroll
  for (int m = 1; m < 16; m <<= 1) x = fmaxf(x, __shfl_xor(x, m, 32));
  return x;
}
__device__ __forceinline__ float red_sum16(float x) {
#pragma unroll
  for (int m = 1; m < 16; m <<= 1) x += __shfl_xor(x, m, 32);
  return x;
}

// ---------------------------------------------------------------------------
// Kernel 0: convert weights to f16 and transpose to [n][k] so B fragments
// load contiguous K pairs.
// ---------------------------------------------------------------------------
__global__ void prep_weights(const float* __restrict__ Wk, const float* __restrict__ Wq,
                             const float* __restrict__ Wv, const float* __restrict__ Wp,
                             _Float16* __restrict__ WkT, _Float16* __restrict__ WqT,
                             _Float16* __restrict__ WvT, _Float16* __restrict__ WpT) {
  const int i = blockIdx.x * blockDim.x + threadIdx.x;
  if (i < CEMB * HEAD) {
    const int k = i / HEAD, n = i % HEAD;
    WkT[n * CEMB + k] = (_Float16)Wk[i];
    WqT[n * CEMB + k] = (_Float16)Wq[i];
    WvT[n * CEMB + k] = (_Float16)Wv[i];
  }
  if (i < HEAD * HEAD) {
    const int k = i / HEAD, n = i % HEAD;
    WpT[n * HEAD + k] = (_Float16)Wp[i];
  }
}

// ---------------------------------------------------------------------------
// Kernel 1: fused Q/K/V projection + value-residual mix.
// One wave computes a 16-row x 64-col tile of q,k,v: 24 K-steps x 12 WMMA.
// ---------------------------------------------------------------------------
__global__ void __launch_bounds__(32)
qkv_kernel(const float* __restrict__ x, const float* __restrict__ v1,
           const _Float16* __restrict__ WkT, const _Float16* __restrict__ WqT,
           const _Float16* __restrict__ WvT, const float* __restrict__ plamb,
           _Float16* __restrict__ qf, _Float16* __restrict__ kf,
           _Float16* __restrict__ vf) {
  const int rowbase = blockIdx.x * 16;
  v8f zero = {};
  v8f cq[4], ck[4], cv[4];
#pragma unroll
  for (int nt = 0; nt < 4; ++nt) { cq[nt] = zero; ck[nt] = zero; cv[nt] = zero; }

  for (int ks = 0; ks < CEMB; ks += 32) {
    v16h ax = frag_rm_f32(x, CEMB, rowbase, ks);
#pragma unroll
    for (int nt = 0; nt < 4; ++nt) {
      cq[nt] = wmma16(ax, frag_rm(WqT, CEMB, nt * 16, ks), cq[nt]);
      ck[nt] = wmma16(ax, frag_rm(WkT, CEMB, nt * 16, ks), ck[nt]);
      cv[nt] = wmma16(ax, frag_rm(WvT, CEMB, nt * 16, ks), cv[nt]);
    }
  }

  const float lam = plamb[0];
  const int lane = threadIdx.x & 31;
  const int r16 = lane & 15, hi = lane >> 4;
#pragma unroll
  for (int nt = 0; nt < 4; ++nt) {
    UV8F q, k, v;
    q.v = cq[nt]; k.v = ck[nt]; v.v = cv[nt];
#pragma unroll
    for (int j = 0; j < 8; ++j) {
      const int row = rowbase + j + 8 * hi;       // C layout: M = j + 8*(lane>=16)
      const int col = nt * 16 + r16;              // N = lane % 16
      const size_t idx = (size_t)row * HEAD + col;
      qf[idx] = (_Float16)q.f[j];
      kf[idx] = (_Float16)k.f[j];
      const float vm = (1.0f - lam) * v.f[j] + lam * v1[idx];
      vf[idx] = (_Float16)vm;
    }
  }
}

// ---------------------------------------------------------------------------
// Kernel 2: flash attention. One wave per 16-query tile; online softmax over
// 32-key chunks; 8 WMMAs per chunk (4 for S = QK^T, 4 for O += P V).
// V chunk streams into LDS via GLOBAL_LOAD_ASYNC_TO_LDS_B128 (ASYNCcnt),
// overlapped with the score WMMAs; waited only before the P*V gathers.
// ---------------------------------------------------------------------------
__global__ void __launch_bounds__(32)
attn_kernel(const _Float16* __restrict__ qf, const _Float16* __restrict__ kf,
            const _Float16* __restrict__ vf, _Float16* __restrict__ of) {
  __shared__ __align__(16) _Float16 ldsV[32 * HEAD];  // V chunk (s-major)
  __shared__ __align__(16) _Float16 ldsP[16 * 32];    // P tile, C->A relayout

  const int tile = blockIdx.x;
  const int b = tile / (SEQ / 16);
  const int qt = tile % (SEQ / 16);
  const int qbase = qt * 16;
  const _Float16* qb = qf + (size_t)b * SEQ * HEAD;
  const _Float16* kb = kf + (size_t)b * SEQ * HEAD;
  const _Float16* vb = vf + (size_t)b * SEQ * HEAD;

  const int lane = threadIdx.x & 31;
  const int r16 = lane & 15, hi = lane >> 4;
  const float NEG = -__builtin_inff();
  const float scale = 0.125f;  // 1/sqrt(64)

  const v16h aq0 = frag_rm(qb, HEAD, qbase, 0);
  const v16h aq1 = frag_rm(qb, HEAD, qbase, 32);

  v8f zero = {};
  v8f o[4];
#pragma unroll
  for (int nt = 0; nt < 4; ++nt) o[nt] = zero;
  float m[8], l[8], alpha[8];
#pragma unroll
  for (int j = 0; j < 8; ++j) { m[j] = NEG; l[j] = 0.0f; }

  const int nchunk = (qbase + 16 + 31) / 32;  // chunks covering cols [0, qbase+15]
  for (int sc = 0; sc < nchunk; ++sc) {
    const int sbase = sc * 32;

    // Previous iteration's ds_load gathers of ldsV/ldsP must complete before
    // this iteration overwrites LDS (the async engine is NOT ordered with the
    // LDS pipeline, so fence DScnt explicitly).
    wait_ds0();
    __syncthreads();

    {  // stage V chunk (32 x 64 f16 = 4 KB)
      const _Float16* src = vb + (size_t)sbase * HEAD;
#if USE_ASYNC_LDS
      as1_i4* g  = (as1_i4*)src;   // addrspace(1) i4*, const dropped (load-only)
      as3_i4* l3 = (as3_i4*)ldsV;  // addrspace(3) i4*
#pragma unroll
      for (int it = 0; it < 8; ++it)
        __builtin_amdgcn_global_load_async_to_lds_b128(g + it * 32 + lane,
                                                       l3 + it * 32 + lane, 0, 0);
#else
      const h8* s8 = (const h8*)src;
      h8* d8 = (h8*)ldsV;
#pragma unroll
      for (int it = 0; it < 8; ++it) d8[it * 32 + lane] = s8[it * 32 + lane];
#endif
    }
    if (sbase + 32 < SEQ) {  // global_prefetch_b8 for next chunk
      __builtin_prefetch(vb + (size_t)(sbase + 32) * HEAD, 0, 1);
      __builtin_prefetch(kb + (size_t)(sbase + 32) * HEAD, 0, 1);
    }

    // S = Q K^T over K=64 (2 WMMAs per 16-col half of the chunk); overlaps
    // with the in-flight async V transfer.
    v8f s0 = wmma16(aq0, frag_rm(kb, HEAD, sbase, 0), zero);
    s0     = wmma16(aq1, frag_rm(kb, HEAD, sbase, 32), s0);
    v8f s1 = wmma16(aq0, frag_rm(kb, HEAD, sbase + 16, 0), zero);
    s1     = wmma16(aq1, frag_rm(kb, HEAD, sbase + 16, 32), s1);

    const bool needmask = (sbase + 31 > qbase);
    UV8F su0, su1, p0, p1;
    su0.v = s0; su1.v = s1;
#pragma unroll
    for (int j = 0; j < 8; ++j) {
      float a0 = su0.f[j] * scale;
      float a1 = su1.f[j] * scale;
      if (needmask) {
        const int rowg = qbase + j + 8 * hi;
        if (sbase + r16 > rowg)      a0 = NEG;
        if (sbase + 16 + r16 > rowg) a1 = NEG;
      }
      float mx = red_max16(fmaxf(a0, a1));
      float mn = fmaxf(m[j], mx);
      float al = __expf(m[j] - mn);
      float e0 = __expf(a0 - mn);
      float e1 = __expf(a1 - mn);
      l[j] = l[j] * al + red_sum16(e0 + e1);
      m[j] = mn;
      alpha[j] = al;
      p0.f[j] = e0;
      p1.f[j] = e1;
    }

    // P: C-layout -> A-fragment via LDS round trip
#pragma unroll
    for (int j = 0; j < 8; ++j) {
      ldsP[(j + 8 * hi) * 32 + r16]      = (_Float16)p0.f[j];
      ldsP[(j + 8 * hi) * 32 + 16 + r16] = (_Float16)p1.f[j];
    }
    __syncthreads();
    const v16h ap = frag_rm((const _Float16*)ldsP, 32, 0, 0);

#if USE_ASYNC_LDS
    wait_async0();  // V chunk now fully in LDS
#endif

    // O = diag(alpha) * O + P V
#pragma unroll
    for (int nt = 0; nt < 4; ++nt) {
      UV8F ou; ou.v = o[nt];
#pragma unroll
      for (int j = 0; j < 8; ++j) ou.f[j] *= alpha[j];
      const v16h bv = frag_colgather(ldsV, HEAD, nt * 16);
      o[nt] = wmma16(ap, bv, ou.v);
    }
  }

  // normalize and store f16 attention output
#pragma unroll
  for (int nt = 0; nt < 4; ++nt) {
    UV8F ou; ou.v = o[nt];
#pragma unroll
    for (int j = 0; j < 8; ++j) {
      const int row = qbase + j + 8 * hi;
      const int col = nt * 16 + r16;
      of[((size_t)b * SEQ + row) * HEAD + col] = (_Float16)(ou.f[j] / l[j]);
    }
  }
}

// ---------------------------------------------------------------------------
// Kernel 3: output projection + bias, f32 result.
// ---------------------------------------------------------------------------
__global__ void __launch_bounds__(32)
proj_kernel(const _Float16* __restrict__ of, const _Float16* __restrict__ WpT,
            const float* __restrict__ bias, float* __restrict__ out) {
  const int rowbase = blockIdx.x * 16;
  const v16h a0 = frag_rm(of, HEAD, rowbase, 0);
  const v16h a1 = frag_rm(of, HEAD, rowbase, 32);
  const int lane = threadIdx.x & 31;
  const int r16 = lane & 15, hi = lane >> 4;
  v8f zero = {};
#pragma unroll
  for (int nt = 0; nt < 4; ++nt) {
    v8f c = wmma16(a0, frag_rm(WpT, HEAD, nt * 16, 0), zero);
    c     = wmma16(a1, frag_rm(WpT, HEAD, nt * 16, 32), c);
    const int col = nt * 16 + r16;
    const float bv = bias[col];
    UV8F cu; cu.v = c;
#pragma unroll
    for (int j = 0; j < 8; ++j)
      out[(size_t)(rowbase + j + 8 * hi) * HEAD + col] = cu.f[j] + bv;
  }
}

// ---------------------------------------------------------------------------
extern "C" void kernel_launch(void* const* d_in, const int* in_sizes, int n_in,
                              void* d_out, int out_size, void* d_ws, size_t ws_size,
                              hipStream_t stream) {
  (void)in_sizes; (void)n_in; (void)out_size; (void)ws_size;
  const float* x     = (const float*)d_in[0];
  const float* v1    = (const float*)d_in[1];
  const float* Wk    = (const float*)d_in[2];
  const float* Wq    = (const float*)d_in[3];
  const float* Wv    = (const float*)d_in[4];
  const float* Wp    = (const float*)d_in[5];
  const float* bproj = (const float*)d_in[6];
  const float* lamb  = (const float*)d_in[7];
  float* out = (float*)d_out;

  // Workspace layout (f16): q, k, v, attn-out, then transposed f16 weights.
  _Float16* qf  = (_Float16*)d_ws;
  _Float16* kf  = qf  + (size_t)BT * HEAD;
  _Float16* vf  = kf  + (size_t)BT * HEAD;
  _Float16* of  = vf  + (size_t)BT * HEAD;
  _Float16* WkT = of  + (size_t)BT * HEAD;
  _Float16* WqT = WkT + CEMB * HEAD;
  _Float16* WvT = WqT + CEMB * HEAD;
  _Float16* WpT = WvT + CEMB * HEAD;  // total ~16.3 MB

  prep_weights<<<(CEMB * HEAD + 255) / 256, 256, 0, stream>>>(
      Wk, Wq, Wv, Wp, WkT, WqT, WvT, WpT);
  qkv_kernel<<<BT / 16, 32, 0, stream>>>(x, v1, WkT, WqT, WvT, lamb, qf, kf, vf);
  attn_kernel<<<BT / 16, 32, 0, stream>>>(qf, kf, vf, of);
  proj_kernel<<<BT / 16, 32, 0, stream>>>(of, WpT, bproj, out);

  // Tuple output: second element is v1 passed through.
  (void)hipMemcpyAsync(out + (size_t)BT * HEAD, v1,
                       (size_t)BT * HEAD * sizeof(float),
                       hipMemcpyDeviceToDevice, stream);
}
